// Block_25262997635346
// MI455X (gfx1250) — matrix operand order repaired and checked by reading, not compiled, not run on hardware
//
#include <hip/hip_runtime.h>
#include <hip/hip_bf16.h>
#include <math.h>
#include <stdint.h>

#define DI __device__ __forceinline__

typedef unsigned int u32;
typedef unsigned short u16;

typedef __attribute__((ext_vector_type(16))) __bf16 v16bf;
typedef __attribute__((ext_vector_type(8)))  float  v8f;
typedef __attribute__((ext_vector_type(4)))  int    v4i;

union FragA { v16bf v; u32 u[8]; };
union FragC { v8f   v; float f[8]; };

// ------------------------------------------------ async LDS staging (CDNA5)
#if defined(__has_builtin)
#if __has_builtin(__builtin_amdgcn_global_load_async_to_lds_b128)
#define HAVE_ASYNC_LDS 1
#endif
#endif

// Copy 16 bytes global -> LDS (per lane). Async DMA when available.
DI void stage16(const u16* g, u16* l) {
#if HAVE_ASYNC_LDS
  __builtin_amdgcn_global_load_async_to_lds_b128((v4i*)g, (v4i*)l, 0, 0);
#else
  const uint4 v = *(const uint4*)g;
  u32* d = (u32*)l;
  d[0] = v.x; d[1] = v.y; d[2] = v.z; d[3] = v.w;
#endif
}
DI void stage_fence() {
#if HAVE_ASYNC_LDS
  asm volatile("s_wait_asynccnt 0" ::: "memory");
#endif
}

// ---------------------------------------------------------------- helpers
DI u16 f2bf(float x) {
  u32 u = __builtin_bit_cast(u32, x);
  u = (u + 0x7FFFu + ((u >> 16) & 1u)) >> 16;   // round-to-nearest-even
  return (u16)u;
}
DI u32 pack2bf(float a, float b) {
  return (u32)f2bf(a) | ((u32)f2bf(b) << 16);
}

DI v8f wmma_bf16(v16bf a, v16bf b, v8f c) {
  return __builtin_amdgcn_wmma_f32_16x16x32_bf16(false, a, false, b, (short)0, c,
                                                 false, false);
}

// A fragment (16x32 bf16, MxK) from LDS stored row-major [rows][STRIDE].
template <int STRIDE>
DI v16bf lds_load_a(const u16* base, int row0, int kstart, int lane) {
  FragA f;
  const u16* p = base + (size_t)(row0 + (lane & 15)) * STRIDE + kstart +
                 ((lane >> 4) << 4);
  const u32* p0 = (const u32*)p;
  const u32* p1 = (const u32*)(p + 8);
#pragma unroll
  for (int v = 0; v < 4; ++v) { f.u[v] = p0[v]; f.u[4 + v] = p1[v]; }
  return f.v;
}

// B fragment (32x16 bf16, KxN) from LDS stored as transposed image [N][STRIDE].
template <int STRIDE>
DI v16bf lds_load_b(const u16* base, int n0, int kstart, int lane) {
  FragA f;
  const u32* p = (const u32*)(base + (size_t)(n0 + (lane & 15)) * STRIDE +
                              kstart + ((lane >> 4) << 4));
#pragma unroll
  for (int v = 0; v < 8; ++v) f.u[v] = p[v];
  return f.v;
}

DI float gelu_f(float x) {
  float x3 = x * x * x;
  return 0.5f * x * (1.0f + tanhf(0.7978845608028654f * (x + 0.044715f * x3)));
}

// ---------------------------------------------------------------- LayerNorm
__global__ __launch_bounds__(256) void ln_bf16_kernel(
    const float* __restrict__ x, const float* __restrict__ w,
    const float* __restrict__ b, u16* __restrict__ out) {
  const int row = blockIdx.x;
  const int t = threadIdx.x;
  const float4 v = ((const float4*)(x + (size_t)row * 1024))[t];
  float s  = v.x + v.y + v.z + v.w;
  float s2 = v.x * v.x + v.y * v.y + v.z * v.z + v.w * v.w;
#pragma unroll
  for (int off = 16; off > 0; off >>= 1) {
    s  += __shfl_xor(s, off, 32);
    s2 += __shfl_xor(s2, off, 32);
  }
  __shared__ float rs[8], rs2[8];
  const int wid = t >> 5, lane = t & 31;
  if (lane == 0) { rs[wid] = s; rs2[wid] = s2; }
  __syncthreads();
  float tot = 0.f, tot2 = 0.f;
#pragma unroll
  for (int i = 0; i < 8; ++i) { tot += rs[i]; tot2 += rs2[i]; }
  const float mean = tot * (1.0f / 1024.0f);
  const float var  = tot2 * (1.0f / 1024.0f) - mean * mean;
  const float rstd = rsqrtf(var + 1e-5f);
  const float4 wv = ((const float4*)w)[t];
  const float4 bv = ((const float4*)b)[t];
  uint2 pk;
  pk.x = pack2bf((v.x - mean) * rstd * wv.x + bv.x,
                 (v.y - mean) * rstd * wv.y + bv.y);
  pk.y = pack2bf((v.z - mean) * rstd * wv.z + bv.z,
                 (v.w - mean) * rstd * wv.w + bv.w);
  ((uint2*)(out + (size_t)row * 1024))[t] = pk;
}

// ---------------------------------------------------------------- GEMM
// C[M,N] = A[M,K](bf16) * W[K,N](f32->bf16) + bias (+GELU)(+residual).
// 128x128 block tile, K-tile 32, double-buffered LDS, one barrier per tile.
// 8 waves in 2(m) x 4(n) grid; each wave: 4x2 WMMA tiles of 16x16.
template <int KTOT, int NTOT, bool OUT_BF16, bool DO_GELU, bool DO_RES>
__global__ __launch_bounds__(256) void gemm_wmma_kernel(
    const u16* __restrict__ A, const float* __restrict__ W,
    const float* __restrict__ bias, const float* __restrict__ res,
    void* __restrict__ outp) {
  __shared__ u16 As[2][128][34];  // [m][k], pad->17-bank stride, conflict free
  __shared__ u16 Bs[2][128][34];  // [n][k] transposed image of W tile

  const int t = threadIdx.x;
  const int lane = t & 31, wid = t >> 5;
  const int n0 = blockIdx.x * 128;
  const int m0 = blockIdx.y * 128;
  const int wm = (wid >> 2) * 64;
  const int wn = (wid & 3) * 32;

  v8f acc[4][2];
#pragma unroll
  for (int i = 0; i < 4; ++i)
#pragma unroll
    for (int j = 0; j < 2; ++j) {
      FragC z;
#pragma unroll
      for (int r = 0; r < 8; ++r) z.f[r] = 0.0f;
      acc[i][j] = z.v;
    }

  const int arow = t >> 1, ahalf = (t & 1) << 4;  // A staging: 2 thr/row
  const int kq4 = (t >> 5) << 2;                  // W staging: 4x4 micro-block
  const int nq4 = (t & 31) << 2;

  // ---- prologue: stage tile 0 into buffer 0
  {
    const u16* gA = A + (size_t)(m0 + arow) * KTOT + ahalf;
    stage16(gA, &As[0][arow][ahalf]);
    stage16(gA + 8, &As[0][arow][ahalf + 8]);
    float wreg[4][4];
#pragma unroll
    for (int i = 0; i < 4; ++i) {
      const float4 f = *(const float4*)(W + (size_t)(kq4 + i) * NTOT + n0 + nq4);
      wreg[i][0] = f.x; wreg[i][1] = f.y; wreg[i][2] = f.z; wreg[i][3] = f.w;
    }
#pragma unroll
    for (int j = 0; j < 4; ++j) {
      u32* d = (u32*)&Bs[0][nq4 + j][kq4];
      d[0] = pack2bf(wreg[0][j], wreg[1][j]);
      d[1] = pack2bf(wreg[2][j], wreg[3][j]);
    }
    stage_fence();
  }
  __syncthreads();

  const int NT = KTOT / 32;
  for (int it = 0; it < NT; ++it) {
    const int cur = it & 1;
    const bool have_next = (it + 1 < NT);
    const int kt_next = (it + 1) * 32;
    float wreg[4][4];
    if (have_next) {
      // issue next A tile (async DMA overlaps with WMMA below)
      const u16* gA = A + (size_t)(m0 + arow) * KTOT + kt_next + ahalf;
      stage16(gA, &As[cur ^ 1][arow][ahalf]);
      stage16(gA + 8, &As[cur ^ 1][arow][ahalf + 8]);
      // issue next W tile global loads (coalesced 512B runs per k-row)
#pragma unroll
      for (int i = 0; i < 4; ++i) {
        const float4 f =
            *(const float4*)(W + (size_t)(kt_next + kq4 + i) * NTOT + n0 + nq4);
        wreg[i][0] = f.x; wreg[i][1] = f.y; wreg[i][2] = f.z; wreg[i][3] = f.w;
      }
      if (it + 2 < NT)
        __builtin_prefetch(W + (size_t)(kt_next + 32 + kq4) * NTOT + n0 + nq4, 0, 1);
    }

    // ---- compute current buffer
    {
      const u16* bs = &Bs[cur][0][0];
      const u16* as = &As[cur][0][0];
      v16bf b0 = lds_load_b<34>(bs, wn, 0, lane);
      v16bf b1 = lds_load_b<34>(bs, wn + 16, 0, lane);
#pragma unroll
      for (int i = 0; i < 4; ++i) {
        v16bf a = lds_load_a<34>(as, wm + 16 * i, 0, lane);
        acc[i][0] = wmma_bf16(a, b0, acc[i][0]);
        acc[i][1] = wmma_bf16(a, b1, acc[i][1]);
      }
    }

    if (have_next) {
      // convert + store next W tile (2 packed b32 stores x4)
#pragma unroll
      for (int j = 0; j < 4; ++j) {
        u32* d = (u32*)&Bs[cur ^ 1][nq4 + j][kq4];
        d[0] = pack2bf(wreg[0][j], wreg[1][j]);
        d[1] = pack2bf(wreg[2][j], wreg[3][j]);
      }
      stage_fence();
    }
    __syncthreads();
  }

  // ---- epilogue
  const int rowhi = (lane & 16) ? 8 : 0;
  const int ncol = lane & 15;
#pragma unroll
  for (int i = 0; i < 4; ++i)
#pragma unroll
    for (int j = 0; j < 2; ++j) {
      FragC c; c.v = acc[i][j];
      const int col = n0 + wn + 16 * j + ncol;
      const float bb = bias[col];
#pragma unroll
      for (int r = 0; r < 8; ++r) {
        const int row = m0 + wm + 16 * i + rowhi + r;
        float vv = c.f[r] + bb;
        if (DO_GELU) vv = gelu_f(vv);
        if (DO_RES) vv += res[(size_t)row * NTOT + col];
        if (OUT_BF16)
          ((u16*)outp)[(size_t)row * NTOT + col] = f2bf(vv);
        else
          ((float*)outp)[(size_t)row * NTOT + col] = vv;
      }
    }
}

// ---------------------------------------------------------------- Attention
// Causal flash attention. Grid: (S/128, B*H); 8 waves, wave w owns q-rows
// w*16..w*16+15 so softmax row stats reduce within 16-lane groups.
__global__ __launch_bounds__(256) void attention_kernel(
    const u16* __restrict__ qkv,   // [B*S, 3*1024] bf16
    u16* __restrict__ attnout) {   // [B*S, 1024] bf16
  __shared__ u16 Qs[128][66];
  __shared__ u16 Ks[64][66];   // [key][d]  == B image for Q·Kᵀ
  __shared__ u16 Vs[64][66];   // [d][key]  == B image for P·V
  __shared__ u16 Ps[128][66];  // per-wave P rows, bf16

  const int t = threadIdx.x;
  const int lane = t & 31, wid = t >> 5;
  const int qt = blockIdx.x;
  const int bh = blockIdx.y;
  const int bi = bh >> 4, hi = bh & 15;
  const size_t rowbase = (size_t)bi * 2048;
  const int qbase = qt * 128;
  const int rowhi = (lane & 16) ? 8 : 0;
  const int ncol = lane & 15;

  {  // stage Q tile: 128 x 64 bf16 (pure copy -> async path)
    const int row = t >> 1, half = (t & 1) << 5;
    const u16* g = qkv + (rowbase + qbase + row) * 3072 + hi * 64 + half;
    u16* l = &Qs[row][half];
    stage16(g, l); stage16(g + 8, l + 8);
    stage16(g + 16, l + 16); stage16(g + 24, l + 24);
    stage_fence();
  }
  __syncthreads();

  FragA qf0, qf1;
  qf0.v = lds_load_a<66>(&Qs[0][0], wid * 16, 0, lane);
  qf1.v = lds_load_a<66>(&Qs[0][0], wid * 16, 32, lane);

  float mrow[8], lrow[8];
  FragC o[4];
#pragma unroll
  for (int r = 0; r < 8; ++r) { mrow[r] = -3.0e38f; lrow[r] = 0.0f; }
#pragma unroll
  for (int f = 0; f < 4; ++f)
#pragma unroll
    for (int r = 0; r < 8; ++r) o[f].f[r] = 0.0f;

  const int ntiles = 2 * (qt + 1);
  const float sc = 0.125f;  // 1/sqrt(64)

  for (int jt = 0; jt < ntiles; ++jt) {
    const int kg = jt * 64;
    __syncthreads();
    {  // stage K (copy) and V (transposed, manual) tiles: 64 x 64 bf16
      const int r = t >> 2, c = (t & 3) << 4;
      const u16* gk = qkv + (rowbase + kg + r) * 3072 + 1024 + hi * 64 + c;
      stage16(gk, &Ks[r][c]); stage16(gk + 8, &Ks[r][c + 8]);
      const uint4* gv =
          (const uint4*)(qkv + (rowbase + kg + r) * 3072 + 2048 + hi * 64 + c);
      uint4 w0 = gv[0], w1 = gv[1];
      const u32 uu[8] = {w0.x, w0.y, w0.z, w0.w, w1.x, w1.y, w1.z, w1.w};
#pragma unroll
      for (int i = 0; i < 8; ++i) {
        Vs[c + 2 * i][r]     = (u16)(uu[i] & 0xFFFFu);
        Vs[c + 2 * i + 1][r] = (u16)(uu[i] >> 16);
      }
      stage_fence();
    }
    __syncthreads();

    // S = Q·Kᵀ
    FragC s[4];
#pragma unroll
    for (int f = 0; f < 4; ++f) {
      FragC z;
#pragma unroll
      for (int r = 0; r < 8; ++r) z.f[r] = 0.0f;
      v16bf b0 = lds_load_b<66>(&Ks[0][0], 16 * f, 0, lane);
      z.v = wmma_bf16(qf0.v, b0, z.v);
      v16bf b1 = lds_load_b<66>(&Ks[0][0], 16 * f, 32, lane);
      z.v = wmma_bf16(qf1.v, b1, z.v);
      s[f] = z;
    }

    // scale + causal mask (masked -> -1e10 exactly, as in reference)
    const bool need_mask = (jt >= 2 * qt);
#pragma unroll
    for (int f = 0; f < 4; ++f) {
      const int col = kg + 16 * f + ncol;
#pragma unroll
      for (int r = 0; r < 8; ++r) {
        const int rowg = qbase + wid * 16 + rowhi + r;
        float vv = s[f].f[r] * sc;
        if (need_mask && col > rowg) vv = -1.0e10f;
        s[f].f[r] = vv;
      }
    }

    // online softmax
    float mt[8];
#pragma unroll
    for (int r = 0; r < 8; ++r)
      mt[r] = fmaxf(fmaxf(s[0].f[r], s[1].f[r]), fmaxf(s[2].f[r], s[3].f[r]));
#pragma unroll
    for (int off = 8; off > 0; off >>= 1)
#pragma unroll
      for (int r = 0; r < 8; ++r)
        mt[r] = fmaxf(mt[r], __shfl_xor(mt[r], off, 32));

    float alpha[8], rsum[8];
#pragma unroll
    for (int r = 0; r < 8; ++r) {
      const float mn = fmaxf(mrow[r], mt[r]);
      alpha[r] = __expf(mrow[r] - mn);
      mrow[r] = mn;
      rsum[r] = 0.0f;
    }
#pragma unroll
    for (int f = 0; f < 4; ++f)
#pragma unroll
      for (int r = 0; r < 8; ++r) {
        const float p = __expf(s[f].f[r] - mrow[r]);
        s[f].f[r] = p;
        rsum[r] += p;
      }
#pragma unroll
    for (int off = 8; off > 0; off >>= 1)
#pragma unroll
      for (int r = 0; r < 8; ++r) rsum[r] += __shfl_xor(rsum[r], off, 32);
#pragma unroll
    for (int r = 0; r < 8; ++r) lrow[r] = lrow[r] * alpha[r] + rsum[r];
#pragma unroll
    for (int f = 0; f < 4; ++f)
#pragma unroll
      for (int r = 0; r < 8; ++r) o[f].f[r] *= alpha[r];

    // accumulator layout -> A-fragment layout via per-wave LDS slice
#pragma unroll
    for (int f = 0; f < 4; ++f)
#pragma unroll
      for (int r = 0; r < 8; ++r)
        Ps[wid * 16 + rowhi + r][16 * f + ncol] = f2bf(s[f].f[r]);
    asm volatile("s_wait_dscnt 0" ::: "memory");  // own-wave LDS RAW

    // O += P·V
#pragma unroll
    for (int ks = 0; ks < 2; ++ks) {
      v16bf pa = lds_load_a<66>(&Ps[0][0], wid * 16, ks * 32, lane);
#pragma unroll
      for (int f = 0; f < 4; ++f) {
        v16bf vb = lds_load_b<66>(&Vs[0][0], 16 * f, ks * 32, lane);
        o[f].v = wmma_bf16(pa, vb, o[f].v);
      }
    }
  }

  // normalize + write merged-heads output (bf16)
#pragma unroll
  for (int f = 0; f < 4; ++f)
#pragma unroll
    for (int r = 0; r < 8; ++r) {
      const size_t row = rowbase + qbase + wid * 16 + rowhi + r;
      const int col = hi * 64 + 16 * f + ncol;
      attnout[row * 1024 + col] = f2bf(o[f].f[r] / lrow[r]);
    }
}

// ---------------------------------------------------------------- launcher
extern "C" void kernel_launch(void* const* d_in, const int* in_sizes, int n_in,
                              void* d_out, int out_size, void* d_ws,
                              size_t ws_size, hipStream_t stream) {
  (void)in_sizes; (void)n_in; (void)out_size; (void)ws_size;
  const float* x       = (const float*)d_in[0];
  const float* ln1_w   = (const float*)d_in[1];
  const float* ln1_b   = (const float*)d_in[2];
  const float* W_attn  = (const float*)d_in[3];
  const float* b_attn  = (const float*)d_in[4];
  const float* W_cproj = (const float*)d_in[5];
  const float* b_cproj = (const float*)d_in[6];
  const float* ln2_w   = (const float*)d_in[7];
  const float* ln2_b   = (const float*)d_in[8];
  const float* W_fc    = (const float*)d_in[9];
  const float* b_fc    = (const float*)d_in[10];
  const float* W_mproj = (const float*)d_in[11];
  const float* b_mproj = (const float*)d_in[12];
  float* out = (float*)d_out;

  char* p = (char*)d_ws;
  u16* h    = (u16*)p;   p += (size_t)8192 * 1024 * 2;
  u16* qkv  = (u16*)p;   p += (size_t)8192 * 3072 * 2;
  u16* att  = (u16*)p;   p += (size_t)8192 * 1024 * 2;
  float* x2 = (float*)p; p += (size_t)8192 * 1024 * 4;
  u16* m    = (u16*)p;   p += (size_t)8192 * 1024 * 2;
  u16* act  = (u16*)p;   p += (size_t)8192 * 4096 * 2;

  ln_bf16_kernel<<<8192, 256, 0, stream>>>(x, ln1_w, ln1_b, h);
  gemm_wmma_kernel<1024, 3072, true, false, false>
      <<<dim3(24, 64), 256, 0, stream>>>(h, W_attn, b_attn, nullptr, (void*)qkv);
  attention_kernel<<<dim3(16, 64), 256, 0, stream>>>(qkv, att);
  gemm_wmma_kernel<1024, 1024, false, false, true>
      <<<dim3(8, 64), 256, 0, stream>>>(att, W_cproj, b_cproj, x, (void*)x2);
  ln_bf16_kernel<<<8192, 256, 0, stream>>>(x2, ln2_w, ln2_b, m);
  gemm_wmma_kernel<1024, 4096, true, true, false>
      <<<dim3(32, 64), 256, 0, stream>>>(m, W_fc, b_fc, nullptr, (void*)act);
  gemm_wmma_kernel<4096, 1024, false, false, true>
      <<<dim3(8, 64), 256, 0, stream>>>(act, W_mproj, b_mproj, x2, (void*)out);
}
